// MultiModalMutualAttention_62637803045235
// MI455X (gfx1250) — compile-verified, matrix operand order checked by reading, not run on hardware
//
#include <hip/hip_runtime.h>
#include <hip/hip_bf16.h>

// ---------------------------------------------------------------------------
// MultiModalMutualAttention for MI455X (gfx1250), wave32 + WMMA bf16.
// B=32, S=197, D=1024, H=16, DK=64.
// All intermediates zero-padded to S=224 (7*32) so every WMMA fragment load is
// unconditional + contiguous (global_load_b128).  The two large GEMMs (proj,
// out) are register-blocked 32x32 per wave: 4 wmma per K-step on 4 fragments.
// ---------------------------------------------------------------------------

#define S_LEN 197
#define SP    224          // padded S: 14 tiles of 16, 7 K-steps of 32
#define STP   14           // SP/16
#define DKV   64
#define DMODEL 1024
#define NBH   512          // B*H

typedef __bf16 bf16_t;
typedef __attribute__((ext_vector_type(16))) __bf16 v16bf;
typedef __attribute__((ext_vector_type(8)))  float  v8f;

// Fragment element e of lane l maps to K = 8*(l/16) + e + (e>=8 ? 8 : 0)
// (ISA 7.12.2, 16-bit A 16x32; B mirrors with N in place of M).

// Unguarded fragment from row-major bf16 matrix (two contiguous 16B chunks).
__device__ __forceinline__ v16bf frag_bf16_rm(const bf16_t* __restrict__ p, int ld,
                                              int row, int k0, int lane) {
  v16bf v;
  const bf16_t* rp = p + (long)row * ld + k0 + ((lane >> 4) << 3);
#pragma unroll
  for (int e = 0; e < 16; ++e) v[e] = rp[e + ((e >= 8) ? 8 : 0)];
  return v;
}

// Unguarded fragment from row-major f32 matrix, converted to bf16 on load.
__device__ __forceinline__ v16bf frag_f32_rm(const float* __restrict__ p, int ld,
                                             int row, int k0, int lane) {
  v16bf v;
  const float* rp = p + (long)row * ld + k0 + ((lane >> 4) << 3);
#pragma unroll
  for (int e = 0; e < 16; ++e) v[e] = (bf16_t)rp[e + ((e >= 8) ? 8 : 0)];
  return v;
}

// Unguarded B fragment where B[k][c] = p[k*ld + c] (k-strided source).
__device__ __forceinline__ v16bf frag_bf16_kstride(const bf16_t* __restrict__ p, int ld,
                                                   int col, int k0, int lane) {
  v16bf v;
  const bf16_t* cp = p + col + (long)(k0 + ((lane >> 4) << 3)) * ld;
#pragma unroll
  for (int e = 0; e < 16; ++e) v[e] = cp[(long)(e + ((e >= 8) ? 8 : 0)) * ld];
  return v;
}

__device__ __forceinline__ v8f wmma_bf16(v16bf a, v16bf b, v8f c) {
  return __builtin_amdgcn_wmma_f32_16x16x32_bf16(false, a, false, b, (short)0, c,
                                                 false, false);
}

// ---------------------------------------------------------------------------
// Zero-fill (uint4 granularity) for K/V padding rows.
// ---------------------------------------------------------------------------
__global__ __launch_bounds__(256) void zero_kernel(uint4* __restrict__ p, long n16) {
  long i = (long)blockIdx.x * blockDim.x + threadIdx.x;
  const long stride = (long)gridDim.x * blockDim.x;
  const uint4 z = make_uint4(0u, 0u, 0u, 0u);
  for (; i < n16; i += stride) p[i] = z;
}

// wo [1024,197] f32 -> zero-padded bf16 [1024,224].  blockIdx.x = row n.
__global__ __launch_bounds__(256) void wopad_kernel(const float* __restrict__ wo,
                                                    bf16_t* __restrict__ wop) {
  const int n = blockIdx.x, k = threadIdx.x;
  if (k < SP)
    wop[(long)n * SP + k] = (k < S_LEN) ? (bf16_t)wo[(long)n * S_LEN + k]
                                        : (bf16_t)0.0f;
}

// ---------------------------------------------------------------------------
// Kernel 1: projection  y = x @ W^T + b  -> bf16 [B,H,SP,DK] (head-split).
//   M = B*S = 6304, N = D = 1024, K = 1024.  32x32 per wave (4 wmma/K-step).
//   Writes only rows s<197; padding rows pre-zeroed by zero_kernel.
// ---------------------------------------------------------------------------
__device__ __forceinline__ void proj_store(const v8f acc, int m0, int n0, int lane,
                                           const float* __restrict__ bias,
                                           bf16_t* __restrict__ y) {
  const int n = n0 + (lane & 15);
  const int h = n >> 6, dk = n & 63;
  const float bv = bias[n];
  const int mbase = m0 + ((lane >> 4) << 3);
  const int bb = mbase / S_LEN;               // single division per store
  const int ss = mbase - bb * S_LEN;
#pragma unroll
  for (int v = 0; v < 8; ++v) {
    int s2 = ss + v, b2 = bb;
    if (s2 >= S_LEN) { s2 -= S_LEN; ++b2; }   // 16-row tile crosses <=1 batch
    y[(((long)b2 * 16 + h) * SP + s2) * DKV + dk] = (bf16_t)(acc[v] + bv);
  }
}

__global__ __launch_bounds__(256) void proj_kernel(const float* __restrict__ x,
                                                   const float* __restrict__ w,
                                                   const float* __restrict__ bias,
                                                   bf16_t* __restrict__ y) {
  const int wave = threadIdx.x >> 5, lane = threadIdx.x & 31;
  const int NT32 = DMODEL / 32;                       // 32
  const long tile = (long)blockIdx.x * 8 + wave;      // 197*32 = 6304 tiles
  const int m0 = (int)(tile / NT32) * 32, n0 = (int)(tile % NT32) * 32;
  const int ar = lane & 15;

  v8f acc00 = {}, acc01 = {}, acc10 = {}, acc11 = {};
  for (int k0 = 0; k0 < DMODEL; k0 += 32) {
    v16bf a0 = frag_f32_rm(x, DMODEL, m0 + ar, k0, lane);
    v16bf a1 = frag_f32_rm(x, DMODEL, m0 + 16 + ar, k0, lane);
    v16bf b0 = frag_f32_rm(w, DMODEL, n0 + ar, k0, lane);   // W^T[k][n]=W[n][k]
    v16bf b1 = frag_f32_rm(w, DMODEL, n0 + 16 + ar, k0, lane);
    acc00 = wmma_bf16(a0, b0, acc00);
    acc01 = wmma_bf16(a0, b1, acc01);
    acc10 = wmma_bf16(a1, b0, acc10);
    acc11 = wmma_bf16(a1, b1, acc11);
  }
  proj_store(acc00, m0,      n0,      lane, bias, y);
  proj_store(acc01, m0,      n0 + 16, lane, bias, y);
  proj_store(acc10, m0 + 16, n0,      lane, bias, y);
  proj_store(acc11, m0 + 16, n0 + 16, lane, bias, y);
}

// ---------------------------------------------------------------------------
// Kernel 2: scores  S01 = K0 @ K1^T * (1/8)  -> f32 into the a0 output region
// ---------------------------------------------------------------------------
__global__ __launch_bounds__(256) void scores_kernel(const bf16_t* __restrict__ k0ws,
                                                     const bf16_t* __restrict__ k1ws,
                                                     float* __restrict__ sbuf) {
  const int bh = blockIdx.x;
  const int wave = threadIdx.x >> 5, lane = threadIdx.x & 31;
  const bf16_t* K0 = k0ws + (long)bh * SP * DKV;
  const bf16_t* K1 = k1ws + (long)bh * SP * DKV;
  float* Sb = sbuf + (long)bh * S_LEN * S_LEN;
  const int T = 13;                                 // 13*16 = 208 covers 197
  const int wslots = gridDim.y * 8;
  for (int t = blockIdx.y * 8 + wave; t < T * T; t += wslots) {
    const int m0 = (t / T) * 16, n0 = (t % T) * 16;
    v8f acc = {};
#pragma unroll
    for (int k0 = 0; k0 < DKV; k0 += 32) {
      v16bf a = frag_bf16_rm(K0, DKV, m0 + (lane & 15), k0, lane);
      v16bf b = frag_bf16_rm(K1, DKV, n0 + (lane & 15), k0, lane);
      acc = wmma_bf16(a, b, acc);
    }
    const int n = n0 + (lane & 15);
    if (n < S_LEN) {
#pragma unroll
      for (int v = 0; v < 8; ++v) {
        int m = m0 + v + ((lane >> 4) << 3);
        if (m < S_LEN) Sb[(long)m * S_LEN + n] = acc[v] * 0.125f;
      }
    }
  }
}

// ---------------------------------------------------------------------------
// Kernel 3: dual softmax.  a0 = row-softmax(S01) in place; since s10 = S01^T,
// a1[t][s] = exp(S[s][t]-cmax[t])/csum[t].  Also writes zero-padded bf16
// copies a0p/a1p [224][224] for the WMMA consumers.
// ---------------------------------------------------------------------------
__global__ __launch_bounds__(256) void softmax_kernel(float* __restrict__ a0,
                                                      float* __restrict__ a1,
                                                      bf16_t* __restrict__ a0pw,
                                                      bf16_t* __restrict__ a1pw) {
  const int bh = blockIdx.x;
  float* Sb = a0 + (long)bh * S_LEN * S_LEN;   // raw scores, becomes a0
  float* A1 = a1 + (long)bh * S_LEN * S_LEN;
  bf16_t* A0p = a0pw + (long)bh * SP * SP;
  bf16_t* A1p = a1pw + (long)bh * SP * SP;
  __shared__ float rmax[S_LEN], rsum[S_LEN], cmax[S_LEN], csum[S_LEN];
  const int tid = threadIdx.x;

  for (int r = tid; r < S_LEN; r += blockDim.x) {
    float mx = -3.0e38f;
    for (int t = 0; t < S_LEN; ++t) mx = fmaxf(mx, Sb[(long)r * S_LEN + t]);
    float sm = 0.0f;
    for (int t = 0; t < S_LEN; ++t) sm += __expf(Sb[(long)r * S_LEN + t] - mx);
    rmax[r] = mx; rsum[r] = sm;
  }
  for (int c = tid; c < S_LEN; c += blockDim.x) {
    float mx = -3.0e38f;
    for (int s = 0; s < S_LEN; ++s) mx = fmaxf(mx, Sb[(long)s * S_LEN + c]);
    float sm = 0.0f;
    for (int s = 0; s < S_LEN; ++s) sm += __expf(Sb[(long)s * S_LEN + c] - mx);
    cmax[c] = mx; csum[c] = sm;
  }
  __syncthreads();
  // Division-free sweep: row loop outer, lanes across columns.
  for (int s = 0; s < SP; ++s) {
    for (int t = tid; t < SP; t += blockDim.x) {
      if (s < S_LEN && t < S_LEN) {
        const float v = Sb[(long)s * S_LEN + t];
        const float a0v = __expf(v - rmax[s]) / rsum[s];
        const float a1v = __expf(v - cmax[t]) / csum[t];
        Sb[(long)s * S_LEN + t] = a0v;           // a0[s][t] in place
        A1[(long)t * S_LEN + s] = a1v;           // a1[t][s]
        A0p[(long)s * SP + t] = (bf16_t)a0v;
        A1p[(long)t * SP + s] = (bf16_t)a1v;
      } else {                                   // zero padding, both copies
        A0p[(long)s * SP + t] = (bf16_t)0.0f;
        A1p[(long)t * SP + s] = (bf16_t)0.0f;    // (s,t)<->(t,s) bijective
      }
    }
  }
}

// ---------------------------------------------------------------------------
// Kernel 4: f_i = a_i @ V_i -> bf16 [SP,64] per (b,h).  Padding self-zeroing:
// a_i padded rows are 0, so padded output rows compute to 0 (stored as-is).
// ---------------------------------------------------------------------------
__global__ __launch_bounds__(256) void av_kernel(const bf16_t* __restrict__ a0pw,
                                                 const bf16_t* __restrict__ a1pw,
                                                 const bf16_t* __restrict__ v0ws,
                                                 const bf16_t* __restrict__ v1ws,
                                                 bf16_t* __restrict__ f0ws,
                                                 bf16_t* __restrict__ f1ws) {
  const int bh = blockIdx.x;
  const int wave = threadIdx.x >> 5, lane = threadIdx.x & 31;
  const int NT = DKV / 16;                 // 4; 14*4 = 56 tiles per output
  const int wslots = gridDim.y * 8;
  for (int j = blockIdx.y * 8 + wave; j < 2 * STP * NT; j += wslots) {
    const int half = (j >= STP * NT);
    const bf16_t* A = (half ? a1pw : a0pw) + (long)bh * SP * SP;
    const bf16_t* V = (half ? v1ws : v0ws) + (long)bh * SP * DKV;
    bf16_t* F = (half ? f1ws : f0ws) + (long)bh * SP * DKV;
    const int tt = j - (half ? STP * NT : 0);
    const int m0 = (tt / NT) * 16, n0 = (tt % NT) * 16;
    v8f acc = {};
#pragma unroll 2
    for (int k0 = 0; k0 < SP; k0 += 32) {
      v16bf a = frag_bf16_rm(A, SP, m0 + (lane & 15), k0, lane);
      v16bf b = frag_bf16_kstride(V, DKV, n0 + (lane & 15), k0, lane);
      acc = wmma_bf16(a, b, acc);
    }
    const int n = n0 + (lane & 15);
#pragma unroll
    for (int v = 0; v < 8; ++v) {
      int m = m0 + v + ((lane >> 4) << 3);
      F[(long)m * DKV + n] = (bf16_t)acc[v];
    }
  }
}

// ---------------------------------------------------------------------------
// Kernel 5: comb = (f0@f1^T + f1@f0^T) * (0.125*0.5), both terms in one acc.
// Padded f rows are zero -> padded comb entries compute to 0 (stored as-is).
// ---------------------------------------------------------------------------
__global__ __launch_bounds__(256) void comb_kernel(const bf16_t* __restrict__ f0ws,
                                                   const bf16_t* __restrict__ f1ws,
                                                   bf16_t* __restrict__ combws) {
  const int bh = blockIdx.x;
  const int wave = threadIdx.x >> 5, lane = threadIdx.x & 31;
  const bf16_t* F0 = f0ws + (long)bh * SP * DKV;
  const bf16_t* F1 = f1ws + (long)bh * SP * DKV;
  bf16_t* C = combws + (long)bh * SP * SP;
  const int wslots = gridDim.y * 8;
  for (int t = blockIdx.y * 8 + wave; t < STP * STP; t += wslots) {
    const int m0 = (t / STP) * 16, n0 = (t % STP) * 16;
    v8f acc = {};
#pragma unroll
    for (int k0 = 0; k0 < DKV; k0 += 32) {
      v16bf a0f = frag_bf16_rm(F0, DKV, m0 + (lane & 15), k0, lane);
      v16bf b1f = frag_bf16_rm(F1, DKV, n0 + (lane & 15), k0, lane);
      acc = wmma_bf16(a0f, b1f, acc);                 // f0 @ f1^T
      v16bf a1f = frag_bf16_rm(F1, DKV, m0 + (lane & 15), k0, lane);
      v16bf b0f = frag_bf16_rm(F0, DKV, n0 + (lane & 15), k0, lane);
      acc = wmma_bf16(a1f, b0f, acc);                 // + f1 @ f0^T
    }
    const int n = n0 + (lane & 15);
#pragma unroll
    for (int v = 0; v < 8; ++v) {
      int m = m0 + v + ((lane >> 4) << 3);
      C[(long)m * SP + n] = (bf16_t)(acc[v] * 0.0625f);
    }
  }
}

// ---------------------------------------------------------------------------
// Kernel 6: out = comb @ wo^T + bo -> f32 [B,H,S,D].  32x32 per wave,
// K padded to 224, wo pre-converted/zero-padded bf16 [1024][224] (L2-resident).
// ---------------------------------------------------------------------------
__global__ __launch_bounds__(256) void out_kernel(const bf16_t* __restrict__ combws,
                                                  const bf16_t* __restrict__ wop,
                                                  const float* __restrict__ bo,
                                                  float* __restrict__ out) {
  const int bh = blockIdx.x;
  const int wave = threadIdx.x >> 5, lane = threadIdx.x & 31;
  const bf16_t* C = combws + (long)bh * SP * SP;
  float* O = out + (long)bh * S_LEN * DMODEL;
  const int MT32 = SP / 32;                // 7
  const int NT32 = DMODEL / 32;            // 32; 7*32 = 224 tiles per (b,h)
  const int ar = lane & 15;
  const int wslots = gridDim.y * 8;
  for (int t = blockIdx.y * 8 + wave; t < MT32 * NT32; t += wslots) {
    const int m0 = (t / NT32) * 32, n0 = (t % NT32) * 32;
    v8f acc00 = {}, acc01 = {}, acc10 = {}, acc11 = {};
#pragma unroll 2
    for (int k0 = 0; k0 < SP; k0 += 32) {
      v16bf a0 = frag_bf16_rm(C, SP, m0 + ar, k0, lane);
      v16bf a1 = frag_bf16_rm(C, SP, m0 + 16 + ar, k0, lane);
      v16bf b0 = frag_bf16_rm(wop, SP, n0 + ar, k0, lane);  // wo^T[k][n]
      v16bf b1 = frag_bf16_rm(wop, SP, n0 + 16 + ar, k0, lane);
      acc00 = wmma_bf16(a0, b0, acc00);
      acc01 = wmma_bf16(a0, b1, acc01);
      acc10 = wmma_bf16(a1, b0, acc10);
      acc11 = wmma_bf16(a1, b1, acc11);
    }
#pragma unroll
    for (int q = 0; q < 4; ++q) {
      const v8f acc = (q == 0) ? acc00 : (q == 1) ? acc01 : (q == 2) ? acc10 : acc11;
      const int mo = m0 + ((q >> 1) << 4);
      const int n  = n0 + ((q & 1) << 4) + ar;
      const float bias = bo[n];
#pragma unroll
      for (int v = 0; v < 8; ++v) {
        int m = mo + v + ((lane >> 4) << 3);
        if (m < S_LEN) O[(long)m * DMODEL + n] = acc[v] + bias;
      }
    }
  }
}

// ---------------------------------------------------------------------------
extern "C" void kernel_launch(void* const* d_in, const int* in_sizes, int n_in,
                              void* d_out, int out_size, void* d_ws, size_t ws_size,
                              hipStream_t stream) {
  const float* x0  = (const float*)d_in[0];
  const float* x1  = (const float*)d_in[1];
  const float* wk0 = (const float*)d_in[2];
  const float* bk0 = (const float*)d_in[3];
  const float* wk1 = (const float*)d_in[4];
  const float* bk1 = (const float*)d_in[5];
  const float* wv0 = (const float*)d_in[6];
  const float* bv0 = (const float*)d_in[7];
  const float* wv1 = (const float*)d_in[8];
  const float* bv1 = (const float*)d_in[9];
  const float* wo  = (const float*)d_in[10];
  const float* bo  = (const float*)d_in[11];

  float* out = (float*)d_out;
  const long OUTN = (long)NBH * S_LEN * DMODEL;   // 103,284,736
  const long AN   = (long)NBH * S_LEN * S_LEN;    //  19,870,208
  float* a0 = out + OUTN;
  float* a1 = a0 + AN;

  bf16_t* ws = (bf16_t*)d_ws;
  const long KVP = (long)NBH * SP * DKV;          //  7,340,032 elems
  const long APN = (long)NBH * SP * SP;           // 25,690,112 elems
  bf16_t* k0b = ws;
  bf16_t* k1b = k0b + KVP;
  bf16_t* v0b = k1b + KVP;
  bf16_t* v1b = v0b + KVP;
  bf16_t* f0b = v1b + KVP;
  bf16_t* f1b = f0b + KVP;
  bf16_t* a0p = f1b + KVP;
  bf16_t* a1p = a0p + APN;
  bf16_t* cmb = a1p + APN;
  bf16_t* wop = cmb + APN;                        // + 1024*224; total ~243 MB

  const dim3 blk(256);
  // Seed K/V padding rows with zeros (58.7 MB, ~2.5 us at 23.3 TB/s).
  zero_kernel<<<2048, blk, 0, stream>>>((uint4*)k0b, (4 * KVP * 2) / 16);
  wopad_kernel<<<DMODEL, blk, 0, stream>>>(wo, wop);

  proj_kernel<<<788, blk, 0, stream>>>(x0, wk0, bk0, k0b);   // 6304/8 tiles
  proj_kernel<<<788, blk, 0, stream>>>(x1, wk1, bk1, k1b);
  proj_kernel<<<788, blk, 0, stream>>>(x0, wv0, bv0, v0b);
  proj_kernel<<<788, blk, 0, stream>>>(x1, wv1, bv1, v1b);
  scores_kernel<<<dim3(NBH, 3), blk, 0, stream>>>(k0b, k1b, a0);
  softmax_kernel<<<NBH, blk, 0, stream>>>(a0, a1, a0p, a1p);
  av_kernel<<<dim3(NBH, 2), blk, 0, stream>>>(a0p, a1p, v0b, v1b, f0b, f1b);
  comb_kernel<<<dim3(NBH, 3), blk, 0, stream>>>(f0b, f1b, cmb);
  out_kernel<<<dim3(NBH, 7), blk, 0, stream>>>(cmb, wop, bo, out);
}
